// NodeRetrieverModel_68788196213329
// MI455X (gfx1250) — matrix operand order repaired and checked by reading, not compiled, not run on hardware
//
#include <hip/hip_runtime.h>
#include <math.h>

// ---------------------------------------------------------------------------
// NodeRetriever fused pipeline for MI455X (gfx1250, wave32, WMMA).
// GEMM [200000,258]x[258,256] via v_wmma_f32_16x16x32_bf16 (f32 acc).
// W is pre-packed once per call into a bf16 fragment-native layout, then
// staged per-block through double-buffered LDS (conflict-free 48B slots),
// so the inner loop is {2x ds_load_b128 + wmma}. Transformed rows are
// reduced in-register (t.q and |t|^2) and never materialized to HBM.
// ---------------------------------------------------------------------------

#define N_NODES 200000
#define DIM     256
#define KTOT    258
#define TOPK    1024
#define NLAB    64
#define EPSF    1e-8f
#define NSTRIPS (N_NODES / 16)        // 12500 row strips of 16

typedef __attribute__((ext_vector_type(16))) __bf16 v16bf;
typedef __attribute__((ext_vector_type(8)))  float  v8f;

// workspace layout in 32-bit words (total ~569k words ~= 2.3 MB)
#define WS_QDEN 0        // f32: max(||q||, eps)
#define WS_M    1        // f32: max logits
#define WS_TBIN 2        // u32: high-16 threshold bin
#define WS_R    3        // u32: remaining count inside high bin
#define WS_KTH  4        // u32: exact 1024-th order key
#define WS_R2   5        // u32: tie budget at K_th
#define WS_TICK 6        // u32: tie ticket counter
#define WS_PMAX 16       // f32[256] per-block max partials
#define WS_PSUM 272      // f32[256] per-block sum-exp partials
#define WS_HHI  1024     // u32[65536] high-16 histogram
#define WS_HLO  (WS_HHI + 65536)   // u32[65536] low-16 histogram
#define WS_S    (WS_HLO + 65536)   // f32[N] logits (t . q)
#define WS_W    (WS_S + N_NODES)   // f32[N] weighted cos (-inf invalid)
#define WS_WPK  (WS_W + N_NODES)   // bf16[9*16*32*16] packed W fragments (36864 words)

// packed-W geometry: fragment (kt,ct) holds, for lane l, the 16 bf16 B values
// B[h] = W[ct*16 + (l&15)][kt*32 + (l>>4)*16 + h]  (zero beyond K=258)
// global: 32B chunk at  kt*16384 + (ct*32+lane)*32
// LDS:    48B slot  at  (ct*32+lane)*48   (padding => conflict-free b128)

__device__ __forceinline__ unsigned short f2bf(float f) {
  unsigned u = __float_as_uint(f);
  u += 0x7FFFu + ((u >> 16) & 1u);            // round-to-nearest-even
  return (unsigned short)(u >> 16);
}
__device__ __forceinline__ unsigned okey(float f) {  // total order on floats
  unsigned u = __float_as_uint(f);
  return (u & 0x80000000u) ? ~u : (u | 0x80000000u);
}

union ABfrag { v16bf v; unsigned short h[16]; uint4 q[2]; };

// --------------------------- kernel 0: ||q|| -------------------------------
__global__ void k_prep(const float* __restrict__ emb,
                       const int* __restrict__ srcp, float* __restrict__ ws) {
  __shared__ float red[256];
  const int t = threadIdx.x;
  const int src = srcp[0];
  const float v = emb[(size_t)src * DIM + t];
  red[t] = v * v;
  __syncthreads();
  for (int o = 128; o > 0; o >>= 1) {
    if (t < o) red[t] += red[t + o];
    __syncthreads();
  }
  if (t == 0) ws[WS_QDEN] = fmaxf(sqrtf(red[0]), EPSF);
}

// ---------------- kernel 0b: pack W into bf16 B-fragment layout ------------
__global__ void k_packw(const float* __restrict__ Wm, unsigned short* __restrict__ wpk) {
  // one thread per (fragment, lane): 9*16 fragments * 32 lanes = 4608 threads
  const int t = blockIdx.x * 256 + threadIdx.x;
  if (t >= 9 * 16 * 32) return;
  const int lane = t & 31;
  const int frag = t >> 5;
  const int kt = frag / 16;
  const int ct = frag % 16;
  const int ncol = ct * 16 + (lane & 15);
  const int kbase = kt * 32 + (lane >> 4) * 16;
  unsigned short* dst = wpk + (size_t)t * 16;
#pragma unroll
  for (int h = 0; h < 16; ++h) {
    const int k = kbase + h;
    dst[h] = (k < KTOT) ? f2bf(Wm[(size_t)ncol * KTOT + k]) : (unsigned short)0;
  }
}

// ------------------ kernel 1: WMMA GEMM + fused row stats ------------------
__global__ void __launch_bounds__(256)
k_gemm(const float* __restrict__ emb, const float* __restrict__ dtab,
       const unsigned short* __restrict__ wpk, const int* __restrict__ dist,
       const int* __restrict__ srcp, float* __restrict__ ws) {
  __shared__ __align__(16) unsigned char bbuf[2][16 * 32 * 48];  // 2 x 24 KB

  const int tid    = threadIdx.x;
  const int lane   = tid & 31;
  const int wave   = tid >> 5;
  const int strip0 = blockIdx.x * 8 + wave;      // 16-row strip index
  const bool active = strip0 < NSTRIPS;
  const int strip  = active ? strip0 : 0;        // tail waves: compute row 0,
                                                 // suppress stores, keep barriers
  const int l16  = lane & 15;
  const int half = lane >> 4;
  const int kg   = half * 16;                    // A fragment K-group offset
  const int rowA = strip * 16 + l16;             // A row owned by this lane
  const float* __restrict__ arow = emb + (size_t)rowA * DIM;
  const unsigned char* __restrict__ wpb = (const unsigned char*)wpk;

  // stage K-slice 0 into LDS buffer 0 (each thread moves two 32B chunks)
  uint4 pr[2][2];
  {
    const uint4* s0 = (const uint4*)(wpb + (size_t)tid * 32);
    const uint4* s1 = (const uint4*)(wpb + (size_t)(tid + 256) * 32);
    pr[0][0] = s0[0]; pr[0][1] = s0[1];
    pr[1][0] = s1[0]; pr[1][1] = s1[1];
    uint4* d0 = (uint4*)(&bbuf[0][(size_t)tid * 48]);
    uint4* d1 = (uint4*)(&bbuf[0][(size_t)(tid + 256) * 48]);
    d0[0] = pr[0][0]; d0[1] = pr[0][1];
    d1[0] = pr[1][0]; d1[1] = pr[1][1];
  }
  __syncthreads();

  v8f zero = {0.f, 0.f, 0.f, 0.f, 0.f, 0.f, 0.f, 0.f};
  v8f acc[16];
#pragma unroll
  for (int ct = 0; ct < 16; ++ct) acc[ct] = zero;

  for (int kt = 0; kt <= 8; ++kt) {
    // kick off next K-slice load from L2 into registers (overlaps WMMAs)
    if (kt < 8) {
      const unsigned char* sb = wpb + (size_t)(kt + 1) * 16384;
      const uint4* s0 = (const uint4*)(sb + (size_t)tid * 32);
      const uint4* s1 = (const uint4*)(sb + (size_t)(tid + 256) * 32);
      pr[0][0] = s0[0]; pr[0][1] = s0[1];
      pr[1][0] = s1[0]; pr[1][1] = s1[1];
    }

    // build A fragment
    ABfrag A;
    if (kt < 8) {                                // K = kt*32 .. kt*32+31 (emb)
      const float* ap = arow + kt * 32 + kg;
      float4 g0 = *(const float4*)(ap);
      float4 g1 = *(const float4*)(ap + 4);
      float4 g2 = *(const float4*)(ap + 8);
      float4 g3 = *(const float4*)(ap + 12);
      if (kt < 7) __builtin_prefetch(arow + (kt + 1) * 32 + kg, 0, 1);
      A.h[0]  = f2bf(g0.x); A.h[1]  = f2bf(g0.y); A.h[2]  = f2bf(g0.z); A.h[3]  = f2bf(g0.w);
      A.h[4]  = f2bf(g1.x); A.h[5]  = f2bf(g1.y); A.h[6]  = f2bf(g1.z); A.h[7]  = f2bf(g1.w);
      A.h[8]  = f2bf(g2.x); A.h[9]  = f2bf(g2.y); A.h[10] = f2bf(g2.z); A.h[11] = f2bf(g2.w);
      A.h[12] = f2bf(g3.x); A.h[13] = f2bf(g3.y); A.h[14] = f2bf(g3.z); A.h[15] = f2bf(g3.w);
    } else {                                     // K = 256..287: distance enc
      const int dv = dist[rowA];
      int di = dv - 1; di = di < 0 ? 0 : (di > 2 ? 2 : di);
      const float d0 = dtab[di * 2 + 0], d1 = dtab[di * 2 + 1];
#pragma unroll
      for (int hh = 0; hh < 16; ++hh) A.h[hh] = 0;
      A.h[0] = (kg == 0) ? f2bf(d0) : (unsigned short)0;
      A.h[1] = (kg == 0) ? f2bf(d1) : (unsigned short)0;
    }

    // 16 column tiles: B fragments from LDS (2x ds_load_b128, conflict-free)
    const unsigned char* bc = bbuf[kt & 1];
#pragma unroll
    for (int ct = 0; ct < 16; ++ct) {
      const uint4* bp = (const uint4*)(bc + (size_t)(ct * 32 + lane) * 48);
      ABfrag B;
      B.q[0] = bp[0];
      B.q[1] = bp[1];
      acc[ct] = __builtin_amdgcn_wmma_f32_16x16x32_bf16(
          false, A.v, false, B.v, (short)0, acc[ct], false, false);
    }

    // commit prefetched slice to the other buffer, sync for next step
    if (kt < 8) {
      uint4* d0 = (uint4*)(&bbuf[(kt + 1) & 1][(size_t)tid * 48]);
      uint4* d1 = (uint4*)(&bbuf[(kt + 1) & 1][(size_t)(tid + 256) * 48]);
      d0[0] = pr[0][0]; d0[1] = pr[0][1];
      d1[0] = pr[1][0]; d1[1] = pr[1][1];
      __syncthreads();
    }
  }

  // fused per-row reductions: s = t.q, n2 = |t|^2 (C layout: VGPR r -> M=r+8*half)
  const int src = srcp[0];
  const float* __restrict__ q = emb + (size_t)src * DIM;
  const float qden = ws[WS_QDEN];

  float sp[8], np[8];
#pragma unroll
  for (int r = 0; r < 8; ++r) { sp[r] = 0.f; np[r] = 0.f; }
#pragma unroll
  for (int ct = 0; ct < 16; ++ct) {
    const float qv = q[ct * 16 + l16];
#pragma unroll
    for (int r = 0; r < 8; ++r) {
      const float v = acc[ct][r];
      sp[r] += v * qv;
      np[r] += v * v;
    }
  }
#pragma unroll
  for (int r = 0; r < 8; ++r) {
#pragma unroll
    for (int off = 1; off < 16; off <<= 1) {     // reduce within 16-lane half
      sp[r] += __shfl_xor(sp[r], off, 32);
      np[r] += __shfl_xor(np[r], off, 32);
    }
  }

  float* __restrict__ s_out = ws + WS_S;
  float* __restrict__ w_out = ws + WS_W;
#pragma unroll
  for (int r = 0; r < 8; ++r) {
    if (active && l16 == r) {
      const int row = strip * 16 + r + 8 * half;
      const float s = sp[r];
      const int dv = dist[row];
      const float wgt = (dv == 1) ? 1.5f : ((dv == 2) ? 2.0f : 1.0f);
      const float cosv = s / (fmaxf(sqrtf(np[r]), EPSF) * qden);
      s_out[row] = s;
      w_out[row] = (dv <= 2) ? cosv * wgt : -INFINITY;
    }
  }
}

// --------------------- logsumexp (deterministic 2-stage) -------------------
__global__ void k_redmax(float* __restrict__ ws) {
  __shared__ float red[256];
  const float* s = ws + WS_S;
  float m = -INFINITY;
  for (int i = blockIdx.x * 256 + threadIdx.x; i < N_NODES; i += 256 * 256)
    m = fmaxf(m, s[i]);
  red[threadIdx.x] = m;
  __syncthreads();
  for (int o = 128; o > 0; o >>= 1) {
    if (threadIdx.x < o) red[threadIdx.x] = fmaxf(red[threadIdx.x], red[threadIdx.x + o]);
    __syncthreads();
  }
  if (threadIdx.x == 0) ws[WS_PMAX + blockIdx.x] = red[0];
}

__global__ void k_maxfin(float* __restrict__ ws) {
  __shared__ float red[256];
  red[threadIdx.x] = ws[WS_PMAX + threadIdx.x];
  __syncthreads();
  for (int o = 128; o > 0; o >>= 1) {
    if (threadIdx.x < o) red[threadIdx.x] = fmaxf(red[threadIdx.x], red[threadIdx.x + o]);
    __syncthreads();
  }
  if (threadIdx.x == 0) ws[WS_M] = red[0];
}

__global__ void k_sumexp(float* __restrict__ ws) {
  __shared__ float red[256];
  const float M = ws[WS_M];
  const float* s = ws + WS_S;
  float a = 0.f;
  for (int i = blockIdx.x * 256 + threadIdx.x; i < N_NODES; i += 256 * 256)
    a += expf(s[i] - M);
  red[threadIdx.x] = a;
  __syncthreads();
  for (int o = 128; o > 0; o >>= 1) {
    if (threadIdx.x < o) red[threadIdx.x] += red[threadIdx.x + o];
    __syncthreads();
  }
  if (threadIdx.x == 0) ws[WS_PSUM + blockIdx.x] = red[0];
}

__global__ void k_loss(const int* __restrict__ labels, float* __restrict__ ws,
                       float* __restrict__ out) {
  __shared__ float red[256];
  const int t = threadIdx.x;
  red[t] = ws[WS_PSUM + t];
  __syncthreads();
  for (int o = 128; o > 0; o >>= 1) {
    if (t < o) red[t] += red[t + o];
    __syncthreads();
  }
  float Sval = red[0];
  __syncthreads();
  red[t] = (t < NLAB) ? ws[WS_S + labels[t]] : 0.f;
  __syncthreads();
  for (int o = 128; o > 0; o >>= 1) {
    if (t < o) red[t] += red[t + o];
    __syncthreads();
  }
  if (t == 0) {
    const float lse = ws[WS_M] + logf(Sval);
    out[0] = (float)NLAB * lse - red[0];
  }
}

// -------------------------- top-k radix select -----------------------------
__global__ void k_zero(unsigned* __restrict__ wsu) {
  const int i = blockIdx.x * 256 + threadIdx.x;
  if (i < 131072) wsu[WS_HHI + i] = 0u;
  if (i == 0) wsu[WS_TICK] = 0u;
}

__global__ void k_histhi(const float* __restrict__ ws, unsigned* __restrict__ wsu) {
  const int i = blockIdx.x * 256 + threadIdx.x;
  if (i < N_NODES) {
    const unsigned k = okey(ws[WS_W + i]);
    atomicAdd(&wsu[WS_HHI + (k >> 16)], 1u);
  }
}

__global__ void k_pickhi(unsigned* __restrict__ wsu) {
  if (threadIdx.x != 0 || blockIdx.x != 0) return;
  unsigned c = 0; int tb = 0; unsigned r = TOPK; bool found = false;
  for (int b = 65535; b >= 0; --b) {
    const unsigned h = wsu[WS_HHI + b];
    if (c + h >= TOPK) { tb = b; r = TOPK - c; found = true; break; }
    c += h;
  }
  if (!found) { tb = 0; r = 0xFFFFFFFFu; }       // fewer than TOPK -> take all
  wsu[WS_TBIN] = (unsigned)tb;
  wsu[WS_R] = r;
}

__global__ void k_histlo(const float* __restrict__ ws, unsigned* __restrict__ wsu) {
  const int i = blockIdx.x * 256 + threadIdx.x;
  if (i < N_NODES) {
    const unsigned k = okey(ws[WS_W + i]);
    if ((k >> 16) == wsu[WS_TBIN]) atomicAdd(&wsu[WS_HLO + (k & 0xFFFFu)], 1u);
  }
}

__global__ void k_picklo(unsigned* __restrict__ wsu) {
  if (threadIdx.x != 0 || blockIdx.x != 0) return;
  const unsigned need = wsu[WS_R];
  const unsigned tb = wsu[WS_TBIN];
  unsigned c = 0; int lb = 0; unsigned r2 = need;
  for (int b = 65535; b >= 0; --b) {
    const unsigned h = wsu[WS_HLO + b];
    if (c + h >= need) { lb = b; r2 = need - c; break; }
    c += h;
  }
  wsu[WS_KTH] = (tb << 16) | (unsigned)lb;
  wsu[WS_R2] = r2;
  wsu[WS_TICK] = 0u;
}

__global__ void k_mask(const float* __restrict__ ws, unsigned* __restrict__ wsu,
                       const int* __restrict__ dist, const int* __restrict__ srcp,
                       float* __restrict__ out) {
  const int i = blockIdx.x * 256 + threadIdx.x;
  if (i >= N_NODES) return;
  const unsigned kth = wsu[WS_KTH];
  const unsigned r2  = wsu[WS_R2];
  const unsigned k = okey(ws[WS_W + i]);
  bool sel = k > kth;
  if (k == kth) sel = atomicAdd(&wsu[WS_TICK], 1u) < r2;
  if (dist[i] == 1 || i == srcp[0]) sel = true;
  out[1 + i] = sel ? 1.0f : 0.0f;
}

// ------------------------------- launcher ----------------------------------
extern "C" void kernel_launch(void* const* d_in, const int* in_sizes, int n_in,
                              void* d_out, int out_size, void* d_ws, size_t ws_size,
                              hipStream_t stream) {
  const float* emb    = (const float*)d_in[0];   // [N, 256]
  const float* dtab   = (const float*)d_in[1];   // [3, 2]
  const float* Wm     = (const float*)d_in[2];   // [256, 258]
  const int*   dist   = (const int*)d_in[3];     // [N]
  const int*   labels = (const int*)d_in[4];     // [64]
  const int*   srcp   = (const int*)d_in[5];     // [1]
  float*    out = (float*)d_out;                 // [1 + N]
  float*    ws  = (float*)d_ws;
  unsigned* wsu = (unsigned*)d_ws;
  unsigned short* wpk = (unsigned short*)(wsu + WS_WPK);

  const int nblk = (N_NODES + 255) / 256;        // 782
  const int gemm_blocks = (NSTRIPS + 7) / 8;     // 1563

  k_zero  <<<512,         256, 0, stream>>>(wsu);
  k_prep  <<<1,           256, 0, stream>>>(emb, srcp, ws);
  k_packw <<<18,          256, 0, stream>>>(Wm, wpk);
  k_gemm  <<<gemm_blocks, 256, 0, stream>>>(emb, dtab, wpk, dist, srcp, ws);
  k_redmax<<<256,         256, 0, stream>>>(ws);
  k_maxfin<<<1,           256, 0, stream>>>(ws);
  k_sumexp<<<256,         256, 0, stream>>>(ws);
  k_loss  <<<1,           256, 0, stream>>>(labels, ws, out);
  k_histhi<<<nblk,        256, 0, stream>>>(ws, wsu);
  k_pickhi<<<1,           1,   0, stream>>>(wsu);
  k_histlo<<<nblk,        256, 0, stream>>>(ws, wsu);
  k_picklo<<<1,           1,   0, stream>>>(wsu);
  k_mask  <<<nblk,        256, 0, stream>>>(ws, wsu, dist, srcp, out);

  (void)in_sizes; (void)n_in; (void)out_size; (void)ws_size;
}
